// MPLayer_51256139710717
// MI455X (gfx1250) — compile-verified, matrix operand order checked by reading, not run on hardware
//
#include <hip/hip_runtime.h>
#include <hip/hip_bf16.h>
#include <stdint.h>

#define DEVI __device__ __forceinline__

typedef __bf16 bf16_t;
typedef __attribute__((ext_vector_type(16))) __bf16 v16bf;
typedef __attribute__((ext_vector_type(8)))  __bf16 v8bf;
typedef __attribute__((ext_vector_type(8)))  float  v8f;

constexpr int kN = 50000;       // atoms
constexpr int kE = 800000;      // edges
constexpr float kEpsBN = 1e-5f;
constexpr float kEpsSM = 1e-16f;

// Branchless, overflow-safe softplus: max(x,0) + log1p(exp(-|x|)).
// exp argument is <= 0, so no overflow guard (and no divergent branch) needed.
DEVI float softplus_f(float x) {
  float e = __expf(-__builtin_fabsf(x));
  return fmaxf(x, 0.f) + __logf(1.f + e);
}
DEVI float silu_f(float x) { return x * (1.f / (1.f + __expf(-x))); }

// Per-lane A-fragment slice of v_wmma_f32_16x16x32_bf16 from a 32-float row:
// elements 0..7 = cols [kbase..kbase+7], elements 8..15 = cols [16+kbase..16+kbase+7].
DEVI v16bf load_row_frag_f32(const float* __restrict__ row, int kbase) {
  const float4 a0 = *(const float4*)(row + kbase);
  const float4 a1 = *(const float4*)(row + kbase + 4);
  const float4 b0 = *(const float4*)(row + 16 + kbase);
  const float4 b1 = *(const float4*)(row + 16 + kbase + 4);
  v16bf r;
  r[0]=(bf16_t)a0.x;  r[1]=(bf16_t)a0.y;  r[2]=(bf16_t)a0.z;  r[3]=(bf16_t)a0.w;
  r[4]=(bf16_t)a1.x;  r[5]=(bf16_t)a1.y;  r[6]=(bf16_t)a1.z;  r[7]=(bf16_t)a1.w;
  r[8]=(bf16_t)b0.x;  r[9]=(bf16_t)b0.y;  r[10]=(bf16_t)b0.z; r[11]=(bf16_t)b0.w;
  r[12]=(bf16_t)b1.x; r[13]=(bf16_t)b1.y; r[14]=(bf16_t)b1.z; r[15]=(bf16_t)b1.w;
  return r;
}

// Same per-lane fragment but from a 32-wide bf16 K-window in LDS.
DEVI v16bf load_lds_frag(const bf16_t* rowk, int kbase) {
  v8bf lo = *(const v8bf*)(rowk + kbase);
  v8bf hi = *(const v8bf*)(rowk + 16 + kbase);
  v16bf r;
#pragma unroll
  for (int i = 0; i < 8; ++i) { r[i] = lo[i]; r[8 + i] = hi[i]; }
  return r;
}

// ---------------------------------------------------------------- K0: zero scratch
__global__ void k_zero(float* __restrict__ p, long long n) {
  long long i = blockIdx.x * (long long)blockDim.x + threadIdx.x;
  for (; i < n; i += (long long)gridDim.x * blockDim.x) p[i] = 0.f;
}

// ---------------------------------------------------------------- K1: GAT edge GEMM
// hi = softplus([a_i|e] @ W), hj = softplus([a_j|e] @ W)  via bf16 WMMA
// alpha_pre = softplus(<hi,att_i> + <hj,att_j>)  (per head, shuffle-reduced)
__global__ void __launch_bounds__(256) k_gat_gemm(
    const float* __restrict__ atom_fea, const long long* __restrict__ edge_idx,
    const float* __restrict__ edge_fea, const float* __restrict__ W,
    const float* __restrict__ att, float* __restrict__ alpha,
    bf16_t* __restrict__ hjbuf) {
  // Pre-swizzled W fragments: [nt(8)][ks(2)][lane(32)][16 bf16]
  __shared__ bf16_t Wf[8 * 2 * 32 * 16];
  for (int s = threadIdx.x; s < 8 * 2 * 32 * 16; s += blockDim.x) {
    int kk = s & 15, lane = (s >> 4) & 31, g = s >> 9;   // g = nt*2+ks
    int nt = g >> 1, ks = g & 1;
    int n  = nt * 16 + (lane & 15);
    int kb = (lane >> 4) * 8;
    int kl = (kk < 8) ? (kb + kk) : (16 + kb + kk - 8);
    Wf[s] = (bf16_t)W[(ks * 32 + kl) * 128 + n];
  }
  __syncthreads();

  const int lane  = threadIdx.x & 31;
  const int wid   = blockIdx.x * (blockDim.x >> 5) + (threadIdx.x >> 5);
  const int nwv   = gridDim.x * (blockDim.x >> 5);
  const int m     = lane & 15;
  const int half  = lane >> 4;
  const int kbase = half * 8;
  const int mbase = half * 8;     // D-matrix M offset for this half
  const int ncol  = m;            // D-matrix N (within 16-wide tile)
  const int ntiles = kE / 16;

  for (int tile = wid; tile < ntiles; tile += nwv) {
    const int e0 = tile * 16;
    const int e  = e0 + m;
    const int ri = (int)edge_idx[e];
    const int ci = (int)edge_idx[kE + e];
    v16bf Aai = load_row_frag_f32(atom_fea + (long long)ri * 32, kbase);
    v16bf Aaj = load_row_frag_f32(atom_fea + (long long)ci * 32, kbase);
    v16bf Aed = load_row_frag_f32(edge_fea + (long long)e * 32, kbase);

    v8f hi[8], hj[8];
#pragma unroll
    for (int nt = 0; nt < 8; ++nt) {
      v8f ai = {}, aj = {};
      v16bf B0 = *(const v16bf*)&Wf[((nt * 2 + 0) * 32 + lane) * 16];
      v16bf B1 = *(const v16bf*)&Wf[((nt * 2 + 1) * 32 + lane) * 16];
      ai = __builtin_amdgcn_wmma_f32_16x16x32_bf16(false, Aai, false, B0, (short)0, ai, false, false);
      ai = __builtin_amdgcn_wmma_f32_16x16x32_bf16(false, Aed, false, B1, (short)0, ai, false, false);
      aj = __builtin_amdgcn_wmma_f32_16x16x32_bf16(false, Aaj, false, B0, (short)0, aj, false, false);
      aj = __builtin_amdgcn_wmma_f32_16x16x32_bf16(false, Aed, false, B1, (short)0, aj, false, false);
#pragma unroll
      for (int r = 0; r < 8; ++r) { ai[r] = softplus_f(ai[r]); aj[r] = softplus_f(aj[r]); }
      hi[nt] = ai; hj[nt] = aj;
    }

    // store hj messages as bf16 [E][128]
#pragma unroll
    for (int nt = 0; nt < 8; ++nt) {
      int nf = nt * 16 + ncol;
#pragma unroll
      for (int r = 0; r < 8; ++r)
        hjbuf[(long long)(e0 + mbase + r) * 128 + nf] = (bf16_t)hj[nt][r];
    }

    // attention logits per head (reduce 32 features = 2 n-tiles x 16 lanes)
#pragma unroll
    for (int h = 0; h < 4; ++h) {
      float p[8];
#pragma unroll
      for (int r = 0; r < 8; ++r) p[r] = 0.f;
#pragma unroll
      for (int q = 0; q < 2; ++q) {
        int nt = 2 * h + q;
        int fih = (nt * 16 + ncol) & 31;
        float wi = att[h * 64 + fih];
        float wj = att[h * 64 + 32 + fih];
#pragma unroll
        for (int r = 0; r < 8; ++r) p[r] += hi[nt][r] * wi + hj[nt][r] * wj;
      }
#pragma unroll
      for (int d = 1; d < 16; d <<= 1) {
#pragma unroll
        for (int r = 0; r < 8; ++r) p[r] += __shfl_xor(p[r], d, 32);
      }
      if (ncol == 0) {
#pragma unroll
        for (int r = 0; r < 8; ++r)
          alpha[(long long)(e0 + mbase + r) * 4 + h] = softplus_f(p[r]);
      }
    }
  }
}

// ---------------------------------------------------------------- K2: BN statistics
__global__ void k_bn_stats(const float* __restrict__ alpha, float* __restrict__ stats) {
  float s[4] = {0, 0, 0, 0}, q[4] = {0, 0, 0, 0};
  for (long long e = blockIdx.x * (long long)blockDim.x + threadIdx.x; e < kE;
       e += (long long)gridDim.x * blockDim.x) {
#pragma unroll
    for (int h = 0; h < 4; ++h) {
      float a = alpha[e * 4 + h];
      s[h] += a; q[h] += a * a;
    }
  }
#pragma unroll
  for (int d = 1; d < 32; d <<= 1) {
#pragma unroll
    for (int h = 0; h < 4; ++h) { s[h] += __shfl_xor(s[h], d, 32); q[h] += __shfl_xor(q[h], d, 32); }
  }
  if ((threadIdx.x & 31) == 0) {
#pragma unroll
    for (int h = 0; h < 4; ++h) { atomicAdd(&stats[h], s[h]); atomicAdd(&stats[4 + h], q[h]); }
  }
}

// ---------------------------------------------------------------- K3: BN apply + segment max
__global__ void k_bn_apply_max(float* __restrict__ alpha, const long long* __restrict__ edge_idx,
                               const float* __restrict__ stats, const float* __restrict__ gamma,
                               const float* __restrict__ beta, float* __restrict__ mbuf) {
  const long long total = (long long)kE * 4;
  for (long long idx = blockIdx.x * (long long)blockDim.x + threadIdx.x; idx < total;
       idx += (long long)gridDim.x * blockDim.x) {
    int h = (int)(idx & 3);
    long long e = idx >> 2;
    float mu  = stats[h] * (1.f / kE);
    float var = stats[4 + h] * (1.f / kE) - mu * mu;
    float y = softplus_f(gamma[h] * (alpha[idx] - mu) * rsqrtf(var + kEpsBN) + beta[h]);
    alpha[idx] = y;                 // y > 0 always, so uint atomicMax is order-preserving
    int r = (int)edge_idx[e];
    atomicMax((unsigned int*)&mbuf[(long long)r * 4 + h], __float_as_uint(y));
  }
}

// ---------------------------------------------------------------- K4: exp + segment sum
__global__ void k_exp_sum(float* __restrict__ alpha, const long long* __restrict__ edge_idx,
                          const float* __restrict__ mbuf, float* __restrict__ sbuf) {
  const long long total = (long long)kE * 4;
  for (long long idx = blockIdx.x * (long long)blockDim.x + threadIdx.x; idx < total;
       idx += (long long)gridDim.x * blockDim.x) {
    int h = (int)(idx & 3);
    long long e = idx >> 2;
    int r = (int)edge_idx[e];
    float ex = __expf(alpha[idx] - mbuf[(long long)r * 4 + h]);
    alpha[idx] = ex;
    atomicAdd(&sbuf[(long long)r * 4 + h], ex);
  }
}

// ---------------------------------------------------------------- K5: weighted scatter-add of messages
__global__ void k_aggregate(const long long* __restrict__ edge_idx, const bf16_t* __restrict__ hjbuf,
                            const float* __restrict__ exbuf, const float* __restrict__ sbuf,
                            float* __restrict__ aggr) {
  const long long total = (long long)kE * 128;
  for (long long idx = blockIdx.x * (long long)blockDim.x + threadIdx.x; idx < total;
       idx += (long long)gridDim.x * blockDim.x) {
    long long e = idx >> 7;
    int f = (int)(idx & 127);
    int h = f >> 5;
    int r = (int)edge_idx[e];
    float w = exbuf[e * 4 + h] / (sbuf[(long long)r * 4 + h] + kEpsSM);
    atomicAdd(&aggr[(long long)r * 128 + f], (float)hjbuf[idx] * w);
  }
}

// ---------------------------------------------------------------- K6: head mean + bias
__global__ void k_node_out(const float* __restrict__ aggr, const float* __restrict__ bias,
                           float* __restrict__ outp) {
  const long long total = (long long)kN * 32;
  for (long long idx = blockIdx.x * (long long)blockDim.x + threadIdx.x; idx < total;
       idx += (long long)gridDim.x * blockDim.x) {
    long long n = idx >> 5;
    int f = (int)(idx & 31);
    const float* a = aggr + n * 128;
    outp[idx] = 0.25f * (a[f] + a[32 + f] + a[64 + f] + a[96 + f]) + bias[f];
  }
}

// ---------------------------------------------------------------- K7: edge MLP via WMMA
// h = silu([out_i|out_j|e] @ W1 + b1) ; edge_out = silu(h @ W2 + b2)
__global__ void __launch_bounds__(128) k_edge_mlp(
    const float* __restrict__ outp, const long long* __restrict__ edge_idx,
    const float* __restrict__ edge_fea,
    const float* __restrict__ W1, const float* __restrict__ b1,
    const float* __restrict__ W2, const float* __restrict__ b2,
    float* __restrict__ edge_out) {
  __shared__ bf16_t W1f[8 * 3 * 32 * 16];    // [nt][ks][lane][16]
  __shared__ bf16_t W2f[2 * 4 * 32 * 16];    // [nt2][ks][lane][16]
  __shared__ bf16_t htile[4][16 * 128];      // per-wave hidden staging
  for (int s = threadIdx.x; s < 8 * 3 * 32 * 16; s += blockDim.x) {
    int kk = s & 15, lane = (s >> 4) & 31, g = s >> 9;   // g = nt*3+ks
    int nt = g / 3, ks = g % 3;
    int n  = nt * 16 + (lane & 15);
    int kb = (lane >> 4) * 8;
    int kl = (kk < 8) ? (kb + kk) : (16 + kb + kk - 8);
    W1f[s] = (bf16_t)W1[(ks * 32 + kl) * 128 + n];
  }
  for (int s = threadIdx.x; s < 2 * 4 * 32 * 16; s += blockDim.x) {
    int kk = s & 15, lane = (s >> 4) & 31, g = s >> 9;   // g = nt2*4+ks
    int nt2 = g >> 2, ks = g & 3;
    int n  = nt2 * 16 + (lane & 15);
    int kb = (lane >> 4) * 8;
    int kl = (kk < 8) ? (kb + kk) : (16 + kb + kk - 8);
    W2f[s] = (bf16_t)W2[(ks * 32 + kl) * 32 + n];
  }
  __syncthreads();

  const int lane  = threadIdx.x & 31;
  const int wslot = threadIdx.x >> 5;
  const int wid   = blockIdx.x * 4 + wslot;
  const int nwv   = gridDim.x * 4;
  const int m     = lane & 15;
  const int half  = lane >> 4;
  const int kbase = half * 8;
  const int mbase = half * 8;
  const int ncol  = m;
  bf16_t* hrow = &htile[wslot][0];
  const int ntiles = kE / 16;

  for (int tile = wid; tile < ntiles; tile += nwv) {
    const int e0 = tile * 16;
    const int e  = e0 + m;
    const int ri = (int)edge_idx[e];
    const int ci = (int)edge_idx[kE + e];
    v16bf A0 = load_row_frag_f32(outp + (long long)ri * 32, kbase);
    v16bf A1 = load_row_frag_f32(outp + (long long)ci * 32, kbase);
    v16bf A2 = load_row_frag_f32(edge_fea + (long long)e * 32, kbase);

#pragma unroll
    for (int nt = 0; nt < 8; ++nt) {
      v8f acc = {};
      v16bf B0 = *(const v16bf*)&W1f[((nt * 3 + 0) * 32 + lane) * 16];
      v16bf B1 = *(const v16bf*)&W1f[((nt * 3 + 1) * 32 + lane) * 16];
      v16bf B2 = *(const v16bf*)&W1f[((nt * 3 + 2) * 32 + lane) * 16];
      acc = __builtin_amdgcn_wmma_f32_16x16x32_bf16(false, A0, false, B0, (short)0, acc, false, false);
      acc = __builtin_amdgcn_wmma_f32_16x16x32_bf16(false, A1, false, B1, (short)0, acc, false, false);
      acc = __builtin_amdgcn_wmma_f32_16x16x32_bf16(false, A2, false, B2, (short)0, acc, false, false);
      int nf = nt * 16 + ncol;
      float bb = b1[nf];
#pragma unroll
      for (int r = 0; r < 8; ++r)
        hrow[(mbase + r) * 128 + nf] = (bf16_t)silu_f(acc[r] + bb);
    }
    // wave-internal producer/consumer through LDS: drain DS pipe before re-reading
    asm volatile("s_wait_dscnt 0" ::: "memory");

    v16bf Ah0 = load_lds_frag(hrow + m * 128 + 0 * 32, kbase);
    v16bf Ah1 = load_lds_frag(hrow + m * 128 + 1 * 32, kbase);
    v16bf Ah2 = load_lds_frag(hrow + m * 128 + 2 * 32, kbase);
    v16bf Ah3 = load_lds_frag(hrow + m * 128 + 3 * 32, kbase);
#pragma unroll
    for (int nt2 = 0; nt2 < 2; ++nt2) {
      v8f acc = {};
      v16bf C0 = *(const v16bf*)&W2f[((nt2 * 4 + 0) * 32 + lane) * 16];
      v16bf C1 = *(const v16bf*)&W2f[((nt2 * 4 + 1) * 32 + lane) * 16];
      v16bf C2 = *(const v16bf*)&W2f[((nt2 * 4 + 2) * 32 + lane) * 16];
      v16bf C3 = *(const v16bf*)&W2f[((nt2 * 4 + 3) * 32 + lane) * 16];
      acc = __builtin_amdgcn_wmma_f32_16x16x32_bf16(false, Ah0, false, C0, (short)0, acc, false, false);
      acc = __builtin_amdgcn_wmma_f32_16x16x32_bf16(false, Ah1, false, C1, (short)0, acc, false, false);
      acc = __builtin_amdgcn_wmma_f32_16x16x32_bf16(false, Ah2, false, C2, (short)0, acc, false, false);
      acc = __builtin_amdgcn_wmma_f32_16x16x32_bf16(false, Ah3, false, C3, (short)0, acc, false, false);
      int nf = nt2 * 16 + ncol;
      float bb = b2[nf];
#pragma unroll
      for (int r = 0; r < 8; ++r)
        edge_out[(long long)(e0 + mbase + r) * 32 + nf] = silu_f(acc[r] + bb);
    }
  }
}

// ---------------------------------------------------------------- launch
extern "C" void kernel_launch(void* const* d_in, const int* in_sizes, int n_in,
                              void* d_out, int out_size, void* d_ws, size_t ws_size,
                              hipStream_t stream) {
  const float*     atom_fea = (const float*)d_in[0];
  const long long* edge_idx = (const long long*)d_in[1];   // jnp.int64
  const float*     edge_fea = (const float*)d_in[2];
  // d_in[3] batch, d_in[4] distance, d_in[5] edge_vec: unused by reference outputs
  const float* W     = (const float*)d_in[6];
  const float* att   = (const float*)d_in[7];
  const float* bias  = (const float*)d_in[8];
  const float* gamma = (const float*)d_in[9];
  const float* beta  = (const float*)d_in[10];
  const float* eW1   = (const float*)d_in[11];
  const float* eb1   = (const float*)d_in[12];
  const float* eW2   = (const float*)d_in[13];
  const float* eb2   = (const float*)d_in[14];

  float* outp     = (float*)d_out;                 // [N,32]
  float* edge_out = outp + (size_t)kN * 32;        // [E,32]

  // workspace layout (all 256B-aligned by construction)
  char* ws = (char*)d_ws;
  float*  alpha = (float*)ws;                                   // [E,4]   12.8 MB
  bf16_t* hjbuf = (bf16_t*)(ws + (size_t)kE * 4 * 4);           // [E,128] 204.8 MB
  float*  zbase = (float*)(ws + (size_t)kE * 4 * 4 + (size_t)kE * 128 * 2);
  float*  stats = zbase;                 // 8
  float*  mbuf  = zbase + 8;             // [N,4]
  float*  sbuf  = mbuf + (size_t)kN * 4; // [N,4]
  float*  aggr  = sbuf + (size_t)kN * 4; // [N,128]
  const long long zcount = 8 + (long long)kN * (4 + 4 + 128);

  k_zero<<<2048, 256, 0, stream>>>(zbase, zcount);

  // 50000 tiles of 16 edges; 8 waves/block -> exactly one tile per wave
  k_gat_gemm<<<6250, 256, 0, stream>>>(atom_fea, edge_idx, edge_fea, W, att, alpha, hjbuf);
  k_bn_stats<<<1024, 256, 0, stream>>>(alpha, stats);
  k_bn_apply_max<<<4096, 256, 0, stream>>>(alpha, edge_idx, stats, gamma, beta, mbuf);
  k_exp_sum<<<4096, 256, 0, stream>>>(alpha, edge_idx, mbuf, sbuf);
  k_aggregate<<<8192, 256, 0, stream>>>(edge_idx, hjbuf, alpha, sbuf, aggr);
  k_node_out<<<3125, 256, 0, stream>>>(aggr, bias, outp);

  // 4 waves/block -> 12500 blocks covers all 50000 tiles
  k_edge_mlp<<<12500, 128, 0, stream>>>(outp, edge_idx, edge_fea, eW1, eb1, eW2, eb2, edge_out);
}
